// Attention_86938728006233
// MI455X (gfx1250) — matrix operand order repaired and checked by reading, not compiled, not run on hardware
//
#include <hip/hip_runtime.h>
#include <hip/hip_bf16.h>

typedef __bf16 bf16;
typedef __bf16 v16bf __attribute__((ext_vector_type(16)));
typedef float  v8f   __attribute__((ext_vector_type(8)));

union FragAB {
    v16bf v;
    uint4 q[2];
};

static constexpr int Bq  = 2;
static constexpr int Lq  = 2048;
static constexpr int Dq  = 2048;
static constexpr int Hq  = 16;
static constexpr int HDq = 128;

// ---------------------------------------------------------------------------
// TDM support (guarded: falls back to vector-load staging if builtin absent)
// ---------------------------------------------------------------------------
#if defined(__gfx1250__) && __has_builtin(__builtin_amdgcn_tensor_load_to_lds) && \
    __has_builtin(__builtin_amdgcn_s_wait_tensorcnt)
#define HAS_TDM 1
#else
#define HAS_TDM 0
#endif

#if HAS_TDM
typedef unsigned int u32x4 __attribute__((ext_vector_type(4)));
typedef int          i32x8 __attribute__((ext_vector_type(8)));
typedef int          i32x4 __attribute__((ext_vector_type(4)));

__device__ inline uint32_t lds_off_of(const void* p) {
    // generic -> LDS address space; ptrtoint yields the 32-bit LDS byte offset
    return (uint32_t)(uintptr_t)(__attribute__((address_space(3))) const char*)p;
}

// 2-D bf16 tile DMA: tileW elems per row (must be 32 -> 64B rows when padded),
// tileH rows, source row stride = strideElems. padded=1 adds 16B of LDS pad
// per 64B row (pad_interval code 3 = 16 DWORDs, pad_amount code 3 = 4 DWORDs),
// matching the 80-byte LDS row stride used by the WMMA fragment readers.
__device__ inline void tdm_load_2d_bf16(const void* gaddr, uint32_t lds_off,
                                        int tileW, int tileH,
                                        int tensorW, int tensorH,
                                        int strideElems, int padded)
{
    const uint64_t ga = (uint64_t)(uintptr_t)gaddr;
    u32x4 g0;
    g0[0] = 1u;                                   // count=1, user descriptor
    g0[1] = lds_off;                              // lds_addr
    g0[2] = (uint32_t)ga;                         // global_addr[31:0]
    g0[3] = (uint32_t)(ga >> 32) | (2u << 30);    // global_addr[56:32] | type=2
    i32x8 g1;
    g1[0] = (1 << 16) |                           // data_size = 2 bytes
            (padded ? ((1 << 20) | (3 << 22) | (3 << 25)) : 0);
    g1[1] = (tensorW & 0xffff) << 16;             // tensor_dim0 low16 @ [63:48]
    g1[2] = ((tensorW >> 16) & 0xffff) |          // tensor_dim0 high16
            ((tensorH & 0xffff) << 16);           // tensor_dim1 low16
    g1[3] = ((tensorH >> 16) & 0xffff) |          // tensor_dim1 high16
            ((tileW & 0xffff) << 16);             // tile_dim0
    g1[4] = (tileH & 0xffff);                     // tile_dim1 (tile_dim2 = 0)
    g1[5] = strideElems;                          // tensor_dim0_stride low32
    g1[6] = 0;
    g1[7] = 0;
    i32x4 z4;
    z4[0] = 0; z4[1] = 0; z4[2] = 0; z4[3] = 0;
    i32x8 z8;
    z8[0] = 0; z8[1] = 0; z8[2] = 0; z8[3] = 0;
    z8[4] = 0; z8[5] = 0; z8[6] = 0; z8[7] = 0;
    // 6-arg form (clang-23 / therock-10.0 headers): groups 2,3 + extra group
    // unused for 2-D tensors, cpol = 0.
    __builtin_amdgcn_tensor_load_to_lds(g0, g1, z4, z4, z8, 0);
}
#endif

// ---------------------------------------------------------------------------
// fp32 -> bf16 bulk converter (8 elems / thread)
// ---------------------------------------------------------------------------
__global__ __launch_bounds__(256)
void cvt_f32_bf16_kernel(const float* __restrict__ src, bf16* __restrict__ dst)
{
    const size_t i = ((size_t)blockIdx.x * 256 + threadIdx.x) * 8;
    const float4 f0 = ((const float4*)(src + i))[0];
    const float4 f1 = ((const float4*)(src + i))[1];
    bf16 t[8];
    t[0] = (bf16)f0.x; t[1] = (bf16)f0.y; t[2] = (bf16)f0.z; t[3] = (bf16)f0.w;
    t[4] = (bf16)f1.x; t[5] = (bf16)f1.y; t[6] = (bf16)f1.z; t[7] = (bf16)f1.w;
    *(uint4*)(dst + i) = *(uint4*)t;
}

// ---------------------------------------------------------------------------
// GEMM: C[M,N] = A[M,K] * Bw[N,K]^T, both operands bf16 in memory.
// 256 threads = 8 waves (4 m x 2 n), block tile 128x128, K-step 32.
// Tiles staged to LDS by TDM (padded rows) or uint4 loads (fallback).
// ---------------------------------------------------------------------------
template<int OutF32>
__global__ __launch_bounds__(256)
void gemm_bf16(const bf16* __restrict__ A, const bf16* __restrict__ Bw,
               void* __restrict__ C_, int M, int N, int K)
{
    // padded row stride: 40 bf16 = 80 bytes (16B-aligned, matches TDM pad)
    __shared__ __align__(16) bf16 As[128 * 40];
    __shared__ __align__(16) bf16 Bs[128 * 40];

    const int tid    = threadIdx.x;
    const int lane   = tid & 31;
    const int laneLo = lane & 15;
    const int laneHi = lane >> 4;
    const int wave   = tid >> 5;
    const int wm     = wave & 3;   // 0..3
    const int wn     = wave >> 2;  // 0..1
    const int bm0    = blockIdx.y * 128;
    const int bn0    = blockIdx.x * 128;

    v8f acc[2][4];
#pragma unroll
    for (int mi = 0; mi < 2; ++mi)
#pragma unroll
        for (int ni = 0; ni < 4; ++ni)
#pragma unroll
            for (int r = 0; r < 8; ++r) acc[mi][ni][r] = 0.0f;

#if HAS_TDM
    const uint32_t asOff = lds_off_of(As);
    const uint32_t bsOff = lds_off_of(Bs);
#else
    const int lrow = tid >> 1;          // 0..127
    const int lk   = (tid & 1) * 16;    // 0 or 16
#endif

    for (int k0 = 0; k0 < K; k0 += 32) {
        __syncthreads();                 // previous iteration's LDS reads done
#if HAS_TDM
        if (wave == 0) {
            tdm_load_2d_bf16(A + (size_t)bm0 * K + k0, asOff,
                             32, 128, K - k0, M - bm0, K, 1);
            tdm_load_2d_bf16(Bw + (size_t)bn0 * K + k0, bsOff,
                             32, 128, K - k0, N - bn0, K, 1);
            __builtin_amdgcn_s_wait_tensorcnt(0);
        }
#else
        {
            const uint4* ap = (const uint4*)(A + (size_t)(bm0 + lrow) * K + k0 + lk);
            *(uint4*)&As[lrow * 40 + lk]     = ap[0];
            *(uint4*)&As[lrow * 40 + lk + 8] = ap[1];
            if (k0 + 32 < K) __builtin_prefetch((const bf16*)ap + 32, 0, 0);
        }
        {
            const uint4* bp = (const uint4*)(Bw + (size_t)(bn0 + lrow) * K + k0 + lk);
            *(uint4*)&Bs[lrow * 40 + lk]     = bp[0];
            *(uint4*)&Bs[lrow * 40 + lk + 8] = bp[1];
            if (k0 + 32 < K) __builtin_prefetch((const bf16*)bp + 32, 0, 0);
        }
#endif
        __syncthreads();

        // ---- fragments from LDS (two ds_load_b128 each) ----
        FragAB af[2], bfr[4];
#pragma unroll
        for (int mi = 0; mi < 2; ++mi) {
            const bf16* p = &As[(wm * 32 + mi * 16 + laneLo) * 40];
            af[mi].q[0] = *(const uint4*)(p + laneHi * 8);
            af[mi].q[1] = *(const uint4*)(p + 16 + laneHi * 8);
        }
#pragma unroll
        for (int ni = 0; ni < 4; ++ni) {
            const bf16* p = &Bs[(wn * 64 + ni * 16 + laneLo) * 40];
            bfr[ni].q[0] = *(const uint4*)(p + laneHi * 8);
            bfr[ni].q[1] = *(const uint4*)(p + 16 + laneHi * 8);
        }
#pragma unroll
        for (int mi = 0; mi < 2; ++mi)
#pragma unroll
            for (int ni = 0; ni < 4; ++ni)
                acc[mi][ni] = __builtin_amdgcn_wmma_f32_16x16x32_bf16(
                    false, af[mi].v, false, bfr[ni].v, (short)0, acc[mi][ni], false, false);
    }

    // ---- epilogue ----
#pragma unroll
    for (int mi = 0; mi < 2; ++mi)
#pragma unroll
        for (int ni = 0; ni < 4; ++ni)
#pragma unroll
            for (int r = 0; r < 8; ++r) {
                const int row = bm0 + wm * 32 + mi * 16 + r + 8 * laneHi;
                const int col = bn0 + wn * 64 + ni * 16 + laneLo;
                if constexpr (OutF32)
                    ((float*)C_)[(size_t)row * N + col] = acc[mi][ni][r];
                else
                    ((bf16*)C_)[(size_t)row * N + col] = (bf16)acc[mi][ni][r];
            }
}

// ---------------------------------------------------------------------------
// RoPE (adjacent-pair rotation over full D) + scatter to [B,H,L,HD] bf16
// ---------------------------------------------------------------------------
__global__ __launch_bounds__(256)
void rope_scatter(const bf16* __restrict__ qkv, const float* __restrict__ cosT,
                  const float* __restrict__ sinT, bf16* __restrict__ Qh,
                  bf16* __restrict__ Kh, bf16* __restrict__ Vh)
{
    const int idx = blockIdx.x * 256 + threadIdx.x;   // pair index
    const int pairsPerTok = 3 * Dq / 2;               // 3072
    const int tok = idx / pairsPerTok;
    const int pr  = idx % pairsPerTok;
    const int b = tok / Lq, l = tok % Lq;
    const int e = pr * 2;
    const int proj = e / Dq;
    const int d = e - proj * Dq;

    const float x1 = (float)qkv[(size_t)tok * 3 * Dq + e];
    const float x2 = (float)qkv[(size_t)tok * 3 * Dq + e + 1];
    float y1 = x1, y2 = x2;
    if (proj < 2) {
        const float c = cosT[l * Dq + d];
        const float s = sinT[l * Dq + d];
        y1 = x1 * c - x2 * s;
        y2 = x2 * c + x1 * s;
    }
    const int h = d / HDq, hd = d % HDq;
    bf16* dst = (proj == 0) ? Qh : (proj == 1) ? Kh : Vh;
    const size_t o = ((size_t)(b * Hq + h) * Lq + l) * HDq + hd;
    dst[o]     = (bf16)y1;
    dst[o + 1] = (bf16)y2;
}

// ---------------------------------------------------------------------------
// Causal flash attention: 4 waves/block, 64 q rows/block, 16 q rows/wave.
// Per 32-key block: 8 WMMA for S=Q*K^T, online softmax, 8 WMMA for P*V.
// ---------------------------------------------------------------------------
__global__ __launch_bounds__(128)
void flash_attn(const bf16* __restrict__ Qh, const bf16* __restrict__ Kb,
                const bf16* __restrict__ Vb, bf16* __restrict__ O)
{
    __shared__ __align__(16) bf16 Vt[128 * 40];    // V^T tile: [hd][k], 10.2 KB
    __shared__ __align__(16) bf16 Pt[4][16 * 40];  // per-wave P tiles, 5.1 KB

    const int tid    = threadIdx.x;
    const int lane   = tid & 31;
    const int laneLo = lane & 15;
    const int laneHi = lane >> 4;
    const int wave   = tid >> 5;

    const int bh   = blockIdx.y;            // 0..B*H-1
    const int b    = bh / Hq, h = bh % Hq;
    const int qblk = blockIdx.x * 64;
    const int q0   = qblk + wave * 16;

    const bf16* Qbase = Qh + (size_t)bh * Lq * HDq;
    const bf16* Kbase = Kb + (size_t)bh * Lq * HDq;
    const bf16* Vbase = Vb + (size_t)bh * Lq * HDq;

    // Q fragments: 4 A-frags (16x32) over HD=128, loaded once.
    FragAB qf[4];
    {
        const bf16* qrow = Qbase + (size_t)(q0 + laneLo) * HDq;
#pragma unroll
        for (int c = 0; c < 4; ++c) {
            qf[c].q[0] = *(const uint4*)(qrow + c * 32 + laneHi * 8);
            qf[c].q[1] = *(const uint4*)(qrow + c * 32 + 16 + laneHi * 8);
        }
    }

    v8f acc[8];
    float m[8], lsum[8];
#pragma unroll
    for (int c = 0; c < 8; ++c)
#pragma unroll
        for (int r = 0; r < 8; ++r) acc[c][r] = 0.0f;
#pragma unroll
    for (int r = 0; r < 8; ++r) { m[r] = -1e30f; lsum[r] = 0.0f; }

    const float scale = 0.08838834764831845f;  // 1/sqrt(128)
    const int kmax = qblk + 64;                // causal bound for this block

    for (int k0 = 0; k0 < kmax; k0 += 32) {
        __syncthreads();
        // ---- stage V^T (32 keys x 128 hd -> Vt[hd][k]) cooperatively ----
        {
            const int r  = tid >> 2;           // key row 0..31
            const int c0 = (tid & 3) * 32;     // hd chunk
            const bf16* vrow = Vbase + (size_t)(k0 + r) * HDq + c0;
            bf16 t[32];
            *(uint4*)&t[0]  = *(const uint4*)(vrow + 0);
            *(uint4*)&t[8]  = *(const uint4*)(vrow + 8);
            *(uint4*)&t[16] = *(const uint4*)(vrow + 16);
            *(uint4*)&t[24] = *(const uint4*)(vrow + 24);
#pragma unroll
            for (int j = 0; j < 32; ++j) Vt[(c0 + j) * 40 + r] = t[j];
        }
        __syncthreads();

        // ---- S = Q * K^T : two 16x16 tiles, K-dim = HD in 4 chunks of 32 ----
        v8f s0, s1;
#pragma unroll
        for (int r = 0; r < 8; ++r) { s0[r] = 0.0f; s1[r] = 0.0f; }
#pragma unroll
        for (int c = 0; c < 4; ++c) {
            FragAB kf;
            const bf16* kr0 = Kbase + (size_t)(k0 + laneLo) * HDq + c * 32;
            kf.q[0] = *(const uint4*)(kr0 + laneHi * 8);
            kf.q[1] = *(const uint4*)(kr0 + 16 + laneHi * 8);
            s0 = __builtin_amdgcn_wmma_f32_16x16x32_bf16(false, qf[c].v, false, kf.v,
                                                         (short)0, s0, false, false);
            const bf16* kr1 = Kbase + (size_t)(k0 + 16 + laneLo) * HDq + c * 32;
            kf.q[0] = *(const uint4*)(kr1 + laneHi * 8);
            kf.q[1] = *(const uint4*)(kr1 + 16 + laneHi * 8);
            s1 = __builtin_amdgcn_wmma_f32_16x16x32_bf16(false, qf[c].v, false, kf.v,
                                                         (short)0, s1, false, false);
        }

        // ---- scale + causal mask ----
        float sv0[8], sv1[8];
#pragma unroll
        for (int r = 0; r < 8; ++r) {
            const int row = q0 + r + 8 * laneHi;
            sv0[r] = (k0 + laneLo <= row)      ? s0[r] * scale : -1e30f;
            sv1[r] = (k0 + 16 + laneLo <= row) ? s1[r] * scale : -1e30f;
        }
        // ---- row max (reduce across lanes 0..15 within each half) ----
        float rm[8];
#pragma unroll
        for (int r = 0; r < 8; ++r) rm[r] = fmaxf(sv0[r], sv1[r]);
#pragma unroll
        for (int msk = 1; msk < 16; msk <<= 1)
#pragma unroll
            for (int r = 0; r < 8; ++r)
                rm[r] = fmaxf(rm[r], __shfl_xor(rm[r], msk, 32));

        // ---- online softmax update ----
        float p0[8], p1[8], corr[8], rs[8];
#pragma unroll
        for (int r = 0; r < 8; ++r) {
            const float nm = fmaxf(m[r], rm[r]);
            corr[r] = __expf(m[r] - nm);
            m[r] = nm;
            p0[r] = __expf(sv0[r] - nm);
            p1[r] = __expf(sv1[r] - nm);
            rs[r] = p0[r] + p1[r];
        }
#pragma unroll
        for (int msk = 1; msk < 16; msk <<= 1)
#pragma unroll
            for (int r = 0; r < 8; ++r)
                rs[r] += __shfl_xor(rs[r], msk, 32);
#pragma unroll
        for (int r = 0; r < 8; ++r) lsum[r] = lsum[r] * corr[r] + rs[r];
#pragma unroll
        for (int c = 0; c < 8; ++c)
#pragma unroll
            for (int r = 0; r < 8; ++r) acc[c][r] *= corr[r];

        // ---- P (C-layout) -> LDS -> A-frag layout (wave-local) ----
        bf16* Pw = &Pt[wave][0];
#pragma unroll
        for (int r = 0; r < 8; ++r) {
            Pw[(r + 8 * laneHi) * 40 + laneLo]      = (bf16)p0[r];
            Pw[(r + 8 * laneHi) * 40 + 16 + laneLo] = (bf16)p1[r];
        }
        asm volatile("s_wait_dscnt 0" ::: "memory");
        FragAB pa;
        {
            const bf16* pr = Pw + laneLo * 40;
            pa.q[0] = *(const uint4*)(pr + laneHi * 8);
            pa.q[1] = *(const uint4*)(pr + 16 + laneHi * 8);
        }

        // ---- acc += P * V (8 hd chunks of 16) ----
#pragma unroll
        for (int c = 0; c < 8; ++c) {
            FragAB vf;
            const bf16* vr = &Vt[(c * 16 + laneLo) * 40];
            vf.q[0] = *(const uint4*)(vr + laneHi * 8);
            vf.q[1] = *(const uint4*)(vr + 16 + laneHi * 8);
            acc[c] = __builtin_amdgcn_wmma_f32_16x16x32_bf16(false, pa.v, false, vf.v,
                                                             (short)0, acc[c], false, false);
        }
    }

    // ---- normalize and store to [B, L, D] bf16 (token-major for out-proj) ----
#pragma unroll
    for (int r = 0; r < 8; ++r) {
        const float inv = 1.0f / lsum[r];
        const int qrow = q0 + r + 8 * laneHi;
        const size_t obase = ((size_t)b * Lq + qrow) * Dq + h * HDq;
#pragma unroll
        for (int c = 0; c < 8; ++c)
            O[obase + c * 16 + laneLo] = (bf16)(acc[c][r] * inv);
    }
}

// ---------------------------------------------------------------------------
extern "C" void kernel_launch(void* const* d_in, const int* in_sizes, int n_in,
                              void* d_out, int out_size, void* d_ws, size_t ws_size,
                              hipStream_t stream)
{
    const float* x    = (const float*)d_in[0];
    const float* cosT = (const float*)d_in[1];
    const float* sinT = (const float*)d_in[2];
    const float* Wqkv = (const float*)d_in[3];
    const float* Wo   = (const float*)d_in[4];
    float* out = (float*)d_out;

    // ---- workspace layout with lifetime aliasing (~109 MB) ----
    char* ws = (char*)d_ws;
    const size_t qkvBytes  = (size_t)Bq * Lq * 3 * Dq * sizeof(bf16);   // 50,331,648
    const size_t xbBytes   = (size_t)Bq * Lq * Dq * sizeof(bf16);       // 16,777,216
    const size_t hBytes    = (size_t)Bq * Hq * Lq * HDq * sizeof(bf16); // 16,777,216

    bf16* qkv   = (bf16*)ws;                      // [0, 50.3M) ; later AO
    char* regR  = ws + qkvBytes;                  // 50.3 MB region, dual-use
    bf16* xb    = (bf16*)regR;                    // live: conv .. gemm1
    bf16* Wqkvb = (bf16*)(regR + xbBytes);        // live: conv .. gemm1
    bf16* Qh    = (bf16*)regR;                    // live: rope .. attn
    bf16* Kh    = (bf16*)(regR + hBytes);
    bf16* Vh    = (bf16*)(regR + 2 * hBytes);
    bf16* Wob   = (bf16*)(ws + qkvBytes + qkvBytes);  // live whole pipeline
    bf16* AO    = qkv;                            // attention out reuses qkv

    // 0) fp32 -> bf16 conversions (one-time, bandwidth-bound)
    cvt_f32_bf16_kernel<<<(Bq * Lq * Dq) / 2048, 256, 0, stream>>>(x, xb);
    cvt_f32_bf16_kernel<<<(3 * Dq * Dq) / 2048, 256, 0, stream>>>(Wqkv, Wqkvb);
    cvt_f32_bf16_kernel<<<(Dq * Dq) / 2048, 256, 0, stream>>>(Wo, Wob);

    // 1) qkv = xb @ Wqkvb^T   (M=4096, N=6144, K=2048) -> bf16
    gemm_bf16<0><<<dim3(6144 / 128, 4096 / 128), 256, 0, stream>>>(
        xb, Wqkvb, qkv, Bq * Lq, 3 * Dq, Dq);

    // 2) RoPE + head scatter -> Q/K/V [B,H,L,HD] bf16 (overwrites xb/Wqkvb)
    rope_scatter<<<(Bq * Lq * 3 * Dq / 2) / 256, 256, 0, stream>>>(
        qkv, cosT, sinT, Qh, Kh, Vh);

    // 3) causal flash attention -> AO [B,L,D] bf16 (reuses qkv buffer)
    flash_attn<<<dim3(Lq / 64, Bq * Hq), 128, 0, stream>>>(Qh, Kh, Vh, AO);

    // 4) out = AO @ Wob^T    (M=4096, N=2048, K=2048) -> fp32
    gemm_bf16<1><<<dim3(2048 / 128, 4096 / 128), 256, 0, stream>>>(
        AO, Wob, out, Bq * Lq, Dq, Dq);
}
